// DifferenceContrastiveLossWithStats_72456098284082
// MI455X (gfx1250) — compile-verified, matrix operand order checked by reading, not run on hardware
//
#include <hip/hip_runtime.h>
#include <hip/hip_bf16.h>

typedef __attribute__((ext_vector_type(16))) _Float16 v16h;
typedef __attribute__((ext_vector_type(8)))  float    v8f;

// Problem geometry (from reference): z is (B=8, D=64, H=128, W=128)
static constexpr int Dc   = 64;
static constexpr int HWc  = 128 * 128;       // H*W (gather stride in floats)
static constexpr float TEMPc = 0.07f;
static constexpr float EPSN  = 1e-12f;
static constexpr int NCHUNK  = 8;            // split column-tile range across blocks

union FragU { v16h h; unsigned int u[8]; };

// Load one 16-lane-striped f16 operand fragment for v_wmma_f32_16x16x32_f16.
// rowPtr = start of this lane's 64-element row (row-major, stride 64 halves).
// Layout per CDNA5 ISA 7.12.2 (16-bit A 16x32): lane L -> M=L%16,
// VGPR v holds K pair k0 = 16*(v>=4) + 8*(L>=16) + 2*(v%4).
// Pairs are contiguous, so each frag lowers to two global_load_b128 per lane.
__device__ __forceinline__ v16h loadFrag(const _Float16* __restrict__ rowPtr, int koff) {
    const int lane = threadIdx.x & 31;
    const int g = (lane >> 4) & 1;
    FragU fu;
#pragma unroll
    for (int v = 0; v < 8; ++v) {
        const int k0 = ((v & 4) << 2) + g * 8 + (v & 3) * 2 + koff;
        unsigned int u;
        __builtin_memcpy(&u, rowPtr + k0, 4);
        fu.u[v] = u;
    }
    return fu.h;
}

__device__ __forceinline__ v8f wmmaAcc(const v16h& a, const v16h& b, const v8f& c) {
    return __builtin_amdgcn_wmma_f32_16x16x32_f16(false, a, false, b, (short)0, c, false, false);
}

// ---------------- gather + L2-normalize -> f16 rows ----------------
__global__ __launch_bounds__(256)
void gather_norm_f16(const float* __restrict__ z, const int* __restrict__ idx,
                     int rows, _Float16* __restrict__ out) {
    int i = blockIdx.x * blockDim.x + threadIdx.x;
    if (i >= rows) return;
    const int n = idx[i];
    const int b = n / HWc, hw = n % HWc;
    const float* base = z + (size_t)b * Dc * HWc + hw;
    float ssq = 0.f;
#pragma unroll 8
    for (int d = 0; d < Dc; ++d) { float v = base[(size_t)d * HWc]; ssq += v * v; }
    const float inv = 1.0f / fmaxf(sqrtf(ssq), EPSN);
    _Float16* o = out + (size_t)i * Dc;
#pragma unroll 8
    for (int d = 0; d < Dc; ++d) o[d] = (_Float16)(base[(size_t)d * HWc] * inv);
}

// gather + normalize -> optional f32 rows, plus column-sum accumulation (LDS staged)
__global__ __launch_bounds__(256)
void gather_norm_f32_sum(const float* __restrict__ z, const int* __restrict__ idx,
                         int rows, float* __restrict__ outRows, float* __restrict__ sumVec) {
    __shared__ float ls[Dc];
    if (threadIdx.x < Dc) ls[threadIdx.x] = 0.f;
    __syncthreads();
    int i = blockIdx.x * blockDim.x + threadIdx.x;
    if (i < rows) {
        const int n = idx[i];
        const int b = n / HWc, hw = n % HWc;
        const float* base = z + (size_t)b * Dc * HWc + hw;
        float ssq = 0.f;
#pragma unroll 8
        for (int d = 0; d < Dc; ++d) { float v = base[(size_t)d * HWc]; ssq += v * v; }
        const float inv = 1.0f / fmaxf(sqrtf(ssq), EPSN);
        for (int dd = 0; dd < Dc; ++dd) {
            const int d = (dd + (int)threadIdx.x) & (Dc - 1);   // stagger LDS atomics
            const float v = base[(size_t)d * HWc] * inv;
            if (outRows) outRows[(size_t)i * Dc + d] = v;
            atomicAdd(&ls[d], v);
        }
    }
    __syncthreads();
    if (threadIdx.x < Dc) atomicAdd(&sumVec[threadIdx.x], ls[threadIdx.x]);
}

// ---------------- main streaming WMMA kernel ----------------
// grid = (A/16 anchor tiles, NCHUNK column chunks); 256 threads = 8 wave32.
// Pass 1 over pos tiles: row-max of raw dots (diagonal [a,a] excluded; only the
// single 16-aligned tile with cBase==tBase can contain diagonal elements).
// Pass 2 over neg tiles: sum of exp2(fma(dot, kA, kB)) == exp(sim - M), M = 1/TEMP.
// Tile index is wave-uniform SGPR (readfirstlane) -> scalar branches. B fragments
// are ping-pong double-buffered with a manually 2x-unrolled loop (no register
// copies) and a peeled tail (no clamped-pointer select); loads for tile i+1 stay
// outstanding across the WMMA + max/exp of tile i.
__global__ __launch_bounds__(256)
void contrastive_main(const _Float16* __restrict__ posN, const _Float16* __restrict__ negN,
                      const int* __restrict__ anchor_idx, int A, int P, int S,
                      float* __restrict__ partMax, float* __restrict__ partSum) {
    const int tile  = blockIdx.x;
    const int chunk = blockIdx.y;
    const int tBase = tile * 16;
    const int lane  = threadIdx.x & 31;
    const int wid   = threadIdx.x >> 5;
    const int n     = lane & 15;
    const int half  = lane >> 4;
    constexpr int STEP = NCHUNK * 8;                       // 64 column tiles
    const size_t strideElems = (size_t)STEP * 16 * Dc;     // halves advanced per step

    __shared__ float sred[8][16];

    // A fragment: anchors = posN rows gathered via anchor_idx (held in VGPRs).
    const int aRow = tBase + n;
    const int pIdx = (aRow < A) ? anchor_idx[aRow] : 0;
    const _Float16* aPtr = posN + (size_t)pIdx * Dc;
    const v16h aLo = loadFrag(aPtr, 0);
    const v16h aHi = loadFrag(aPtr, 32);

    const float invT = 1.0f / TEMPc;
    const float M    = invT;                               // unit dots -> sim <= 1/TEMP
    const float kA   = invT * 1.44269504088896340736f;     // log2(e)/TEMP
    const float kB   = -M * 1.44269504088896340736f;

    const int ct0 = __builtin_amdgcn_readfirstlane(chunk * 8 + wid);  // wave-uniform

    // ---- pass 1: positives, masked row max (raw-dot domain) ----
    float mx[8];
#pragma unroll
    for (int v = 0; v < 8; ++v) mx[v] = -__builtin_inff();
    {
        const int tiles = (P + 15) >> 4;
        auto consume = [&](int cBase, const v8f& c) {
            if (cBase == tBase || cBase + 16 > P) {        // scalar branch (rare)
                const int col = cBase + n;
#pragma unroll
                for (int v = 0; v < 8; ++v) {
                    const int row = tBase + v + 8 * half;  // global anchor index a
                    if (col < P && col != row) mx[v] = fmaxf(mx[v], c[v]);
                }
            } else {
#pragma unroll
                for (int v = 0; v < 8; ++v) mx[v] = fmaxf(mx[v], c[v]);
            }
        };
        int ct = ct0;
        if (ct < tiles) {
            const _Float16* row = posN + ((size_t)ct * 16 + n) * Dc;
            v16h L0 = loadFrag(row, 0), H0 = loadFrag(row, 32);
            for (;;) {
                int ct1 = ct + STEP;
                if (ct1 >= tiles) {                        // tail in buffer 0
                    v8f c = {}; c = wmmaAcc(aLo, L0, c); c = wmmaAcc(aHi, H0, c);
                    consume(ct * 16, c); break;
                }
                row += strideElems;
                __builtin_prefetch(row + strideElems, 0, 1);
                const v16h L1 = loadFrag(row, 0), H1 = loadFrag(row, 32);
                { v8f c = {}; c = wmmaAcc(aLo, L0, c); c = wmmaAcc(aHi, H0, c);
                  consume(ct * 16, c); }
                ct = ct1; ct1 = ct + STEP;
                if (ct1 >= tiles) {                        // tail in buffer 1
                    v8f c = {}; c = wmmaAcc(aLo, L1, c); c = wmmaAcc(aHi, H1, c);
                    consume(ct * 16, c); break;
                }
                row += strideElems;
                __builtin_prefetch(row + strideElems, 0, 1);
                L0 = loadFrag(row, 0); H0 = loadFrag(row, 32);
                { v8f c = {}; c = wmmaAcc(aLo, L1, c); c = wmmaAcc(aHi, H1, c);
                  consume(ct * 16, c); }
                ct = ct1;
            }
        }
    }
#pragma unroll
    for (int v = 0; v < 8; ++v)
        for (int msk = 8; msk >= 1; msk >>= 1)
            mx[v] = fmaxf(mx[v], __shfl_xor(mx[v], msk, 16));
    if (n == 0) {
#pragma unroll
        for (int v = 0; v < 8; ++v) sred[wid][v + 8 * half] = mx[v];
    }
    __syncthreads();
    if (threadIdx.x < 16) {
        float v = sred[0][threadIdx.x];
#pragma unroll
        for (int w = 1; w < 8; ++w) v = fmaxf(v, sred[w][threadIdx.x]);
        partMax[((size_t)tile * NCHUNK + chunk) * 16 + threadIdx.x] = v * invT;
    }
    __syncthreads();

    // ---- pass 2: negatives, sum exp(sim - M) ----
    float sm[8];
#pragma unroll
    for (int v = 0; v < 8; ++v) sm[v] = 0.f;
    {
        const int tiles = (S + 15) >> 4;
        auto consume = [&](int cBase, const v8f& c) {
            if (cBase + 16 > S) {                          // scalar branch (rare)
                const bool colOK = (cBase + n) < S;
#pragma unroll
                for (int v = 0; v < 8; ++v) {
                    const float e = __builtin_amdgcn_exp2f(fmaf(c[v], kA, kB));
                    sm[v] += colOK ? e : 0.f;
                }
            } else {
#pragma unroll
                for (int v = 0; v < 8; ++v)
                    sm[v] += __builtin_amdgcn_exp2f(fmaf(c[v], kA, kB));
            }
        };
        int ct = ct0;
        if (ct < tiles) {
            const _Float16* row = negN + ((size_t)ct * 16 + n) * Dc;
            v16h L0 = loadFrag(row, 0), H0 = loadFrag(row, 32);
            for (;;) {
                int ct1 = ct + STEP;
                if (ct1 >= tiles) {                        // tail in buffer 0
                    v8f c = {}; c = wmmaAcc(aLo, L0, c); c = wmmaAcc(aHi, H0, c);
                    consume(ct * 16, c); break;
                }
                row += strideElems;
                __builtin_prefetch(row + strideElems, 0, 1);
                const v16h L1 = loadFrag(row, 0), H1 = loadFrag(row, 32);
                { v8f c = {}; c = wmmaAcc(aLo, L0, c); c = wmmaAcc(aHi, H0, c);
                  consume(ct * 16, c); }
                ct = ct1; ct1 = ct + STEP;
                if (ct1 >= tiles) {                        // tail in buffer 1
                    v8f c = {}; c = wmmaAcc(aLo, L1, c); c = wmmaAcc(aHi, H1, c);
                    consume(ct * 16, c); break;
                }
                row += strideElems;
                __builtin_prefetch(row + strideElems, 0, 1);
                L0 = loadFrag(row, 0); H0 = loadFrag(row, 32);
                { v8f c = {}; c = wmmaAcc(aLo, L1, c); c = wmmaAcc(aHi, H1, c);
                  consume(ct * 16, c); }
                ct = ct1;
            }
        }
    }
#pragma unroll
    for (int v = 0; v < 8; ++v)
        for (int msk = 8; msk >= 1; msk >>= 1)
            sm[v] += __shfl_xor(sm[v], msk, 16);
    if (n == 0) {
#pragma unroll
        for (int v = 0; v < 8; ++v) sred[wid][v + 8 * half] = sm[v];
    }
    __syncthreads();
    if (threadIdx.x < 16) {
        float v = 0.f;
#pragma unroll
        for (int w = 0; w < 8; ++w) v += sred[w][threadIdx.x];
        partSum[((size_t)tile * NCHUNK + chunk) * 16 + threadIdx.x] = v;
    }
}

__global__ __launch_bounds__(256)
void finalize_loss(const float* __restrict__ partMax, const float* __restrict__ partSum,
                   int A, float* __restrict__ accum) {
    int a = blockIdx.x * blockDim.x + threadIdx.x;
    if (a >= A) return;
    const int tile = a >> 4, r = a & 15;
    float mx = -__builtin_inff(), s = 0.f;
#pragma unroll
    for (int c = 0; c < NCHUNK; ++c) {
        mx = fmaxf(mx, partMax[((size_t)tile * NCHUNK + c) * 16 + r]);
        s += partSum[((size_t)tile * NCHUNK + c) * 16 + r];
    }
    const float M = 1.0f / TEMPc;
    const float lse = M + logf(__expf(mx - M) + s);
    atomicAdd(&accum[0], lse - mx);
}

// per_pos[i] = (ap_i . sum(ap) - ap_i.ap_i)/(nap-1); per_neg[i] = ap_i . sum(an)/nan
__global__ __launch_bounds__(256)
void accuracy_kernel(const float* __restrict__ apN, const float* __restrict__ apSum,
                     const float* __restrict__ anSum, int nap, int nan,
                     float* __restrict__ accum) {
    const int wid  = (blockIdx.x * blockDim.x + threadIdx.x) >> 5;  // one wave per row
    const int lane = threadIdx.x & 31;
    if (wid >= nap) return;
    const float* row = apN + (size_t)wid * Dc;
    const float d0 = row[lane], d1 = row[lane + 32];
    float dotS    = d0 * apSum[lane] + d1 * apSum[lane + 32];
    float dotSelf = d0 * d0 + d1 * d1;
    float dotAn   = d0 * anSum[lane] + d1 * anSum[lane + 32];
    for (int m = 16; m >= 1; m >>= 1) {
        dotS    += __shfl_xor(dotS, m, 32);
        dotSelf += __shfl_xor(dotSelf, m, 32);
        dotAn   += __shfl_xor(dotAn, m, 32);
    }
    if (lane == 0) {
        const float per_pos = (dotS - dotSelf) / (float)(nap - 1);
        const float per_neg = dotAn / (float)nan;
        if (per_pos > per_neg) atomicAdd(&accum[1], 1.0f);
    }
}

__global__ __launch_bounds__(256)
void mask_count(const float* __restrict__ mask, int N, float* __restrict__ accum) {
    __shared__ float ls;
    if (threadIdx.x == 0) ls = 0.f;
    __syncthreads();
    float cnt = 0.f;
    for (int i = blockIdx.x * blockDim.x + threadIdx.x; i < N; i += gridDim.x * blockDim.x)
        cnt += (mask[i] > 0.5f) ? 1.f : 0.f;
    for (int m = 16; m >= 1; m >>= 1) cnt += __shfl_xor(cnt, m, 32);
    if ((threadIdx.x & 31) == 0) atomicAdd(&ls, cnt);
    __syncthreads();
    if (threadIdx.x == 0) atomicAdd(&accum[2], ls);
}

__global__ __launch_bounds__(256)
void init_ws(float* __restrict__ apSum, float* __restrict__ anSum, float* __restrict__ accum,
             float* __restrict__ partMax, float* __restrict__ partSum, int partN) {
    int i = blockIdx.x * blockDim.x + threadIdx.x;
    if (i < Dc) { apSum[i] = 0.f; anSum[i] = 0.f; }
    if (i < 4) accum[i] = 0.f;
    if (i < partN) { partMax[i] = -__builtin_inff(); partSum[i] = 0.f; }
}

__global__ void finalize_out(const float* __restrict__ accum, int A, int nap, int N,
                             float* __restrict__ out) {
    if (threadIdx.x == 0 && blockIdx.x == 0) {
        out[0] = accum[0] / (float)A;
        out[1] = accum[1] / (float)nap;
        out[2] = accum[2] / (float)N;
    }
}

extern "C" void kernel_launch(void* const* d_in, const int* in_sizes, int n_in,
                              void* d_out, int out_size, void* d_ws, size_t ws_size,
                              hipStream_t stream) {
    const float* z          = (const float*)d_in[0];
    const float* mask       = (const float*)d_in[1];
    const int*   pos_idx    = (const int*)d_in[2];
    const int*   anchor_idx = (const int*)d_in[3];
    const int*   neg_idx    = (const int*)d_in[4];
    const int*   acc_pos    = (const int*)d_in[5];
    const int*   acc_neg    = (const int*)d_in[6];
    float* out = (float*)d_out;

    const int N    = in_sizes[1];
    const int P    = in_sizes[2];
    const int A    = in_sizes[3];
    const int S    = in_sizes[4];
    const int nap  = in_sizes[5];
    const int nan_ = in_sizes[6];
    const int ATILES = (A + 15) / 16;
    const int partN  = ATILES * NCHUNK * 16;

    // workspace layout (256B aligned)
    auto alignup = [](size_t x) { return (x + 255) & ~(size_t)255; };
    char* ws = (char*)d_ws;
    size_t o = 0;
    _Float16* posN = (_Float16*)(ws + o); o = alignup(o + (size_t)P * Dc * 2);
    _Float16* negN = (_Float16*)(ws + o); o = alignup(o + (size_t)S * Dc * 2);
    float* apN     = (float*)(ws + o);    o = alignup(o + (size_t)nap * Dc * 4);
    float* apSum   = (float*)(ws + o);    o = alignup(o + Dc * 4);
    float* anSum   = (float*)(ws + o);    o = alignup(o + Dc * 4);
    float* accum   = (float*)(ws + o);    o = alignup(o + 4 * 4);
    float* partMax = (float*)(ws + o);    o = alignup(o + (size_t)partN * 4);
    float* partSum = (float*)(ws + o);    o = alignup(o + (size_t)partN * 4);

    init_ws<<<(partN + 255) / 256, 256, 0, stream>>>(apSum, anSum, accum, partMax, partSum, partN);

    gather_norm_f16<<<(P + 255) / 256, 256, 0, stream>>>(z, pos_idx, P, posN);
    gather_norm_f16<<<(S + 255) / 256, 256, 0, stream>>>(z, neg_idx, S, negN);
    gather_norm_f32_sum<<<(nap + 255) / 256, 256, 0, stream>>>(z, acc_pos, nap, apN, apSum);
    gather_norm_f32_sum<<<(nan_ + 255) / 256, 256, 0, stream>>>(z, acc_neg, nan_, nullptr, anSum);

    dim3 grid(ATILES, NCHUNK);
    contrastive_main<<<grid, 256, 0, stream>>>(posN, negN, anchor_idx, A, P, S, partMax, partSum);

    finalize_loss<<<(A + 255) / 256, 256, 0, stream>>>(partMax, partSum, A, accum);
    accuracy_kernel<<<(nap * 32 + 255) / 256, 256, 0, stream>>>(apN, apSum, anSum, nap, nan_, accum);
    mask_count<<<256, 256, 0, stream>>>(mask, N, accum);
    finalize_out<<<1, 64, 0, stream>>>(accum, A, nap, N, out);
}
